// MultiHeadAttention_28226525069706
// MI455X (gfx1250) — matrix operand order repaired and checked by reading, not compiled, not run on hardware
//
#include <hip/hip_runtime.h>
#include <hip/hip_bf16.h>

typedef __bf16 bf16;
typedef bf16  v16bf  __attribute__((ext_vector_type(16)));
typedef bf16  bf16x8 __attribute__((ext_vector_type(8)));
typedef float v8f    __attribute__((ext_vector_type(8)));

#if defined(__AMDGCN__) && __has_builtin(__builtin_amdgcn_global_load_async_to_lds_b128)
#define HAVE_ASYNC_LDS 1
typedef int v4i __attribute__((ext_vector_type(4)));
typedef __attribute__((address_space(1))) v4i as1_v4i;  // device global
typedef __attribute__((address_space(3))) v4i as3_v4i;  // LDS
#else
#define HAVE_ASYNC_LDS 0
#endif

// 16-byte global -> LDS copy: async on gfx1250 (ASYNCcnt), sync fallback otherwise
__device__ __forceinline__ void copy16_to_lds(const bf16* __restrict__ gp, bf16* lp) {
#if HAVE_ASYNC_LDS
  __builtin_amdgcn_global_load_async_to_lds_b128((as1_v4i*)gp, (as3_v4i*)lp, 0, 0);
#else
  *(bf16x8*)lp = *(const bf16x8*)gp;
#endif
}

__device__ __forceinline__ void wait_async_lds() {
#if HAVE_ASYNC_LDS
#if __has_builtin(__builtin_amdgcn_s_wait_asynccnt)
  __builtin_amdgcn_s_wait_asynccnt(0);
#else
  asm volatile("s_wait_asynccnt 0x0" ::: "memory");
#endif
#endif
}

__device__ __forceinline__ void store_c(float* p, float v) { *p = v; }
__device__ __forceinline__ void store_c(bf16*  p, float v) { *p = (bf16)v; }

// ----- tile staging: ROWS x 32 operand panel into padded LDS (stride 40 elems) -----
template<int ROWS, int NT>
__device__ __forceinline__ void stage(const float* __restrict__ g, long ld, int k0,
                                      bf16 (* __restrict__ lds)[40], int tid) {
  // f32 source: load + convert to bf16 through VGPRs (ds_store path)
  for (int i = tid; i < ROWS * 8; i += NT) {
    int row = i >> 3, c4 = (i & 7) * 4;
    float4 v = *(const float4*)&g[row * ld + k0 + c4];
    bf16* d = &lds[row][c4];
    d[0] = (bf16)v.x; d[1] = (bf16)v.y; d[2] = (bf16)v.z; d[3] = (bf16)v.w;
  }
}
template<int ROWS, int NT>
__device__ __forceinline__ void stage(const bf16* __restrict__ g, long ld, int k0,
                                      bf16 (* __restrict__ lds)[40], int tid) {
  // bf16 source: direct global->LDS 16B chunks (async on gfx1250)
  for (int i = tid; i < ROWS * 4; i += NT) {
    int row = i >> 2, q = (i & 3) * 8;
    copy16_to_lds(&g[row * ld + k0 + q], &lds[row][q]);
  }
}

// ---------------- generic WMMA GEMM: C = alpha * A(MxK) * B(NxK)^T --------------
// A row-major lda, B row-major (N rows of K) ldb, C row-major ldc.
// Batch: blockIdx.z -> (zb = z>>4, zh = z&15) with per-operand strides.
// Wave tile 32x64 (2x4 subtiles of 16x16); block tile (32*WR) x (64*WC).
// Double-buffered LDS: stage k+32 while computing k.
template<typename TA, typename TB, typename TC, int WR, int WC>
__global__ __launch_bounds__(WR * WC * 32)
void wmma_gemm_kernel(const TA* __restrict__ A, const TB* __restrict__ B,
                      TC* __restrict__ C, int K,
                      long lda, long ldb, long ldc,
                      long sAb, long sAh, long sBb, long sBh, long sCb, long sCh,
                      float alpha) {
  constexpr int BM = 32 * WR, BN = 64 * WC, NT = WR * WC * 32;
  __shared__ __align__(16) bf16 As[2][BM][40];  // 40-elem rows: 80B, 16B aligned
  __shared__ __align__(16) bf16 Bs[2][BN][40];

  const int zb = blockIdx.z >> 4, zh = blockIdx.z & 15;
  const long m0 = (long)blockIdx.y * BM;
  const long n0 = (long)blockIdx.x * BN;
  const TA* Ap = A + (long)zb * sAb + (long)zh * sAh + m0 * lda;
  const TB* Bp = B + (long)zb * sBb + (long)zh * sBh + n0 * ldb;
  TC*       Cb = C + (long)zb * sCb + (long)zh * sCh;

  const int tid  = threadIdx.x;
  const int wave = tid >> 5, lane = tid & 31;
  const int g = lane >> 4, r = lane & 15;   // lane half-group, row-in-group
  const int wm = (wave / WC) * 32;
  const int wn = (wave % WC) * 64;

  v8f acc[2][4];
#pragma unroll
  for (int i = 0; i < 2; ++i)
#pragma unroll
    for (int j = 0; j < 4; ++j)
#pragma unroll
      for (int e = 0; e < 8; ++e) acc[i][j][e] = 0.0f;

  // prologue: stage first K-panel into buffer 0
  stage<BM, NT>(Ap, lda, 0, As[0], tid);
  stage<BN, NT>(Bp, ldb, 0, Bs[0], tid);

  int buf = 0;
  for (int k0 = 0; k0 < K; k0 += 32) {
    wait_async_lds();       // our async chunks for buf are in LDS
    __syncthreads();        // all waves' stage (ds or async) visible

    if (k0 + 32 < K) {      // issue next panel into the other buffer
      stage<BM, NT>(Ap, lda, k0 + 32, As[buf ^ 1], tid);
      stage<BN, NT>(Bp, ldb, k0 + 32, Bs[buf ^ 1], tid);
    }

    // fragments per ISA 7.12.2 (16-bit A 16x32 / B 32x16 layouts)
    v16bf af[2], bfg[4];
#pragma unroll
    for (int i = 0; i < 2; ++i) {
      const bf16* pa = &As[buf][wm + i * 16 + r][0];
      bf16x8 lo = *(const bf16x8*)&pa[8 * g];        // K = 8g..8g+7
      bf16x8 hi = *(const bf16x8*)&pa[16 + 8 * g];   // K = 16+8g..+7
#pragma unroll
      for (int e = 0; e < 8; ++e) { af[i][e] = lo[e]; af[i][8 + e] = hi[e]; }
    }
#pragma unroll
    for (int j = 0; j < 4; ++j) {
      const bf16* pb = &Bs[buf][wn + j * 16 + r][0];
      bf16x8 lo = *(const bf16x8*)&pb[8 * g];        // K = 8g..8g+7
      bf16x8 hi = *(const bf16x8*)&pb[16 + 8 * g];   // K = 16+8g..+7
#pragma unroll
      for (int e = 0; e < 8; ++e) { bfg[j][e] = lo[e]; bfg[j][8 + e] = hi[e]; }
    }

#pragma unroll
    for (int i = 0; i < 2; ++i)
#pragma unroll
      for (int j = 0; j < 4; ++j)
        acc[i][j] = __builtin_amdgcn_wmma_f32_16x16x32_bf16(
            false, af[i], false, bfg[j], (short)0, acc[i][j], false, false);

    buf ^= 1;
  }

  // store: C/D layout — VGPR v holds M = 8*g + v, N = r (per 16x16 subtile)
#pragma unroll
  for (int i = 0; i < 2; ++i)
#pragma unroll
    for (int j = 0; j < 4; ++j)
#pragma unroll
      for (int v = 0; v < 8; ++v) {
        long row = m0 + wm + i * 16 + g * 8 + v;
        long col = n0 + wn + j * 16 + r;
        store_c(&Cb[row * ldc + col], acc[i][j][v] * alpha);
      }
}

// ---------------- V transpose: (b,t,h*64+d) -> (b,h,d,t), bf16 ----------------
__global__ __launch_bounds__(256)
void transpose_v_kernel(const bf16* __restrict__ Vp, bf16* __restrict__ Vt) {
  __shared__ bf16 tile[32][33];
  const int b = blockIdx.z >> 4, h = blockIdx.z & 15;
  const int t0 = blockIdx.x * 32, d0 = blockIdx.y * 32;
  const int tx = threadIdx.x, ty = threadIdx.y;  // (32, 8)
#pragma unroll
  for (int i = 0; i < 32; i += 8)
    tile[ty + i][tx] = Vp[((long)(b * 2048 + t0 + ty + i)) * 1024 + h * 64 + d0 + tx];
  __syncthreads();
#pragma unroll
  for (int i = 0; i < 32; i += 8)
    Vt[((long)blockIdx.z * 64 + d0 + ty + i) * 2048 + t0 + tx] = tile[tx][ty + i];
}

// ---------------- row softmax over T=2048, in place (f32) ----------------
__global__ __launch_bounds__(256)
void softmax_rows_2048(float* __restrict__ attn) {
  const int T = 2048;
  float* p = attn + (long)blockIdx.x * T;
  const int tid = threadIdx.x;
  __shared__ float red[256];

  float loc[8];
  float vmax = -3.0e38f;
#pragma unroll
  for (int i = 0; i < 8; ++i) { loc[i] = p[tid + i * 256]; vmax = fmaxf(vmax, loc[i]); }
  red[tid] = vmax; __syncthreads();
  for (int s = 128; s > 0; s >>= 1) {
    if (tid < s) red[tid] = fmaxf(red[tid], red[tid + s]);
    __syncthreads();
  }
  vmax = red[0]; __syncthreads();

  float sum = 0.0f;
#pragma unroll
  for (int i = 0; i < 8; ++i) { loc[i] = __expf(loc[i] - vmax); sum += loc[i]; }
  red[tid] = sum; __syncthreads();
  for (int s = 128; s > 0; s >>= 1) {
    if (tid < s) red[tid] += red[tid + s];
    __syncthreads();
  }
  const float inv = 1.0f / red[0];
#pragma unroll
  for (int i = 0; i < 8; ++i) p[tid + i * 256] = loc[i] * inv;
}

// ---------------- launcher ----------------
extern "C" void kernel_launch(void* const* d_in, const int* in_sizes, int n_in,
                              void* d_out, int out_size, void* d_ws, size_t ws_size,
                              hipStream_t stream) {
  (void)in_sizes; (void)n_in; (void)out_size; (void)ws_size;
  const float* q    = (const float*)d_in[0];
  const float* k    = (const float*)d_in[1];
  const float* v    = (const float*)d_in[2];
  const float* w_qs = (const float*)d_in[3];
  const float* w_ks = (const float*)d_in[4];
  const float* w_vs = (const float*)d_in[5];
  const float* w_fc = (const float*)d_in[6];

  const int B = 4, T = 2048, D = 1024, H = 16, DK = 64;
  const long MT = (long)B * T;  // 8192 rows

  float* out  = (float*)d_out;
  float* attn = out + MT * D;   // second tuple output, (B,H,T,T) f32

  bf16* Qp = (bf16*)d_ws;       // (B*T, 1024) bf16, Q pre-scaled by 1/8
  bf16* Kp = Qp + MT * D;
  bf16* Vp = Kp + MT * D;
  bf16* Vt = Vp + MT * D;       // (B,H,64,T) bf16
  bf16* Cx = Vt + MT * D;       // attention context (B*T, 1024) bf16

  // 1) projections: (8192x1024) = X(8192x1024) @ W(1024x1024)^T
  dim3 gp(D / 128, MT / 128, 1);
  wmma_gemm_kernel<float, float, bf16, 4, 2><<<gp, 256, 0, stream>>>(
      q, w_qs, Qp, D, D, D, D, 0, 0, 0, 0, 0, 0, 0.125f);  // fold 1/sqrt(64)
  wmma_gemm_kernel<float, float, bf16, 4, 2><<<gp, 256, 0, stream>>>(
      k, w_ks, Kp, D, D, D, D, 0, 0, 0, 0, 0, 0, 1.0f);
  wmma_gemm_kernel<float, float, bf16, 4, 2><<<gp, 256, 0, stream>>>(
      v, w_vs, Vp, D, D, D, D, 0, 0, 0, 0, 0, 0, 1.0f);

  // 2) transpose V to (b,h,d,t)
  transpose_v_kernel<<<dim3(T / 32, DK / 32, B * H), dim3(32, 8), 0, stream>>>(Vp, Vt);

  // 3) scores per (b,h): (2048x2048) = Qh(2048x64) @ Kh(2048x64)^T (bf16 async staging)
  wmma_gemm_kernel<bf16, bf16, float, 4, 2><<<dim3(T / 128, T / 128, B * H), 256, 0, stream>>>(
      Qp, Kp, attn, DK, D, D, T,
      (long)T * D, (long)DK, (long)T * D, (long)DK,
      (long)H * T * T, (long)T * T, 1.0f);

  // 4) softmax rows, in place (final attn output)
  softmax_rows_2048<<<B * H * T, 256, 0, stream>>>(attn);

  // 5) context per (b,h): (2048x64) = attn @ Vh; Vt rows are d-major (bf16 async staging)
  wmma_gemm_kernel<float, bf16, bf16, 4, 1><<<dim3(1, T / 128, B * H), 128, 0, stream>>>(
      attn, Vt, Cx, T, T, T, D,
      (long)H * T * T, (long)T * T, (long)H * DK * T, (long)DK * T,
      (long)T * D, (long)DK, 1.0f);

  // 6) output projection: out(8192x1024) = Cx @ w_fc^T, f32
  wmma_gemm_kernel<bf16, float, float, 4, 2><<<gp, 256, 0, stream>>>(
      Cx, w_fc, out, D, D, D, D, 0, 0, 0, 0, 0, 0, 1.0f);
}
